// RNN_12154757448342
// MI455X (gfx1250) — compile-verified, matrix operand order checked by reading, not run on hardware
//
#include <hip/hip_runtime.h>

#define DEVFN __device__ __forceinline__

typedef __attribute__((ext_vector_type(16))) __bf16 v16bf;
typedef __attribute__((ext_vector_type(8)))  __bf16 v8bf;
typedef __attribute__((ext_vector_type(8)))  float  v8f;

constexpr int Bc = 64;    // batch
constexpr int Tc = 1024;  // time
constexpr int Ic = 256;   // input dim
constexpr int Hc = 512;   // hidden dim

// ---------------------------------------------------------------- helpers
DEVFN __bf16 f32_to_bf16(float f) {
  unsigned int u = __float_as_uint(f);
  u += 0x7FFFu + ((u >> 16) & 1u);   // round-to-nearest-even
  unsigned short hs = (unsigned short)(u >> 16);
  return __builtin_bit_cast(__bf16, hs);
}

DEVFN float sigmoidf_(float x) { return 1.0f / (1.0f + expf(-x)); }

DEVFN v8f wmma_bf16(v16bf a, v16bf b, v8f c) {
  // D = A(16x32 bf16) * B(32x16 bf16) + C(16x16 f32)
  return __builtin_amdgcn_wmma_f32_16x16x32_bf16(
      /*neg_a=*/false, a, /*neg_b=*/false, b,
      /*c_mod=*/(short)0, c, /*reuse_a=*/false, /*reuse_b=*/false);
}

// A fragment, 16-bit 16x32 layout (ISA 7.12.2):
// lane L: row m = m0 + L%16, half = L/16.
// elems 0..7  -> K = k0 + half*8 + i        (contiguous 16B)
// elems 8..15 -> K = k0 + 16 + half*8 + i-8 (contiguous 16B)
DEVFN v16bf load_a_frag(const __bf16* __restrict__ base, int row_stride,
                        int m0, int k0, int lane) {
  const int m    = m0 + (lane & 15);
  const int half = lane >> 4;
  const __bf16* p = base + (size_t)m * row_stride + k0 + half * 8;
  v8bf lo = *(const v8bf*)(p);
  v8bf hi = *(const v8bf*)(p + 16);
  v16bf r;
#pragma unroll
  for (int i = 0; i < 8; ++i) { r[i] = lo[i]; r[i + 8] = hi[i]; }
  return r;
}

// B fragment from pre-packed weights: contiguous 16 bf16 (32B) per lane.
DEVFN v16bf load_b_frag(const __bf16* __restrict__ packed, int kc,
                        int n_tile, int n_tiles, int lane) {
  const __bf16* p = packed + ((((size_t)kc * n_tiles + n_tile) * 32) + lane) * 16;
  return *(const v16bf*)p;
}

// ---------------------------------------------------------------- prep kernels
__global__ void cvt_f32_bf16(const float* __restrict__ src,
                             __bf16* __restrict__ dst, size_t n) {
  size_t i = (size_t)blockIdx.x * blockDim.x + threadIdx.x;
  if (i < n) dst[i] = f32_to_bf16(src[i]);
}

// Pack W[Kdim,H] (row-major f32) into B-fragment order (bf16):
// out[(((kc*n_tiles + n_tile)*32 + lane)*16 + i]
//   = W[kc*32 + kofs(i, lane/16)][n_tile*16 + lane%16]
__global__ void pack_w(const float* __restrict__ W, __bf16* __restrict__ out,
                       int Kdim, int Hdim) {
  size_t idx = (size_t)blockIdx.x * blockDim.x + threadIdx.x;
  size_t total = (size_t)Kdim * Hdim;
  if (idx >= total) return;
  int i      = (int)(idx & 15);
  size_t r   = idx >> 4;
  int lane   = (int)(r & 31); r >>= 5;
  int n_tiles = Hdim / 16;
  int n_tile = (int)(r % n_tiles);
  int kc     = (int)(r / n_tiles);
  int half   = lane >> 4;
  int n = n_tile * 16 + (lane & 15);
  int k = kc * 32 + ((i < 8) ? (half * 8 + i) : (16 + half * 8 + (i - 8)));
  out[idx] = f32_to_bf16(W[(size_t)k * Hdim + n]);
}

__global__ void init_h(float* __restrict__ hf, __bf16* __restrict__ hb, int n) {
  int i = blockIdx.x * blockDim.x + threadIdx.x;
  if (i < n) { hf[i] = 0.0f; hb[i] = f32_to_bf16(0.0f); }
}

// ---------------------------------------------------------------- GRU step
// One launch per time step. 32 blocks x 128 threads = 128 waves; each wave
// owns a 16(M batch)x16(N hidden) tile and fuses:
//   acc_r  = x_t@W_ir + h@W_hr
//   acc_z  = x_t@W_iz + h@W_hz
//   acc_xn = x_t@W_in          acc_hn = h@W_hn   (split for the r* term)
__global__ __launch_bounds__(128) void gru_step(
    const __bf16* __restrict__ x_bf,
    const __bf16* __restrict__ Wir_p, const __bf16* __restrict__ Wiz_p,
    const __bf16* __restrict__ Win_p,
    const __bf16* __restrict__ Whr_p, const __bf16* __restrict__ Whz_p,
    const __bf16* __restrict__ Whn_p,
    const float* __restrict__ b_ir, const float* __restrict__ b_iz,
    const float* __restrict__ b_in, const float* __restrict__ b_hn,
    const float* __restrict__ h_in, const __bf16* __restrict__ hb_in,
    float* __restrict__ h_out, __bf16* __restrict__ hb_out,
    float* __restrict__ y, int t) {
  const int lane   = threadIdx.x & 31;
  const int w      = blockIdx.x * 4 + (threadIdx.x >> 5);  // 0..127
  const int m0     = (w >> 5) << 4;                        // batch tile
  const int n_tile = w & 31;
  const int n0     = n_tile << 4;
  const int n_tiles = Hc / 16;

  v8f acc_r = {}, acc_z = {}, acc_xn = {}, acc_hn = {};

  // input-projection part: A = x[:, t, :], K = Ic
  const __bf16* xrow = x_bf + (size_t)t * Ic;
#pragma unroll 2
  for (int kc = 0; kc < Ic / 32; ++kc) {
    v16bf a  = load_a_frag(xrow, Tc * Ic, m0, kc * 32, lane);
    v16bf br = load_b_frag(Wir_p, kc, n_tile, n_tiles, lane);
    v16bf bz = load_b_frag(Wiz_p, kc, n_tile, n_tiles, lane);
    v16bf bn = load_b_frag(Win_p, kc, n_tile, n_tiles, lane);
    acc_r  = wmma_bf16(a, br, acc_r);
    acc_z  = wmma_bf16(a, bz, acc_z);
    acc_xn = wmma_bf16(a, bn, acc_xn);
  }

  // recurrent part: A = h (bf16), K = Hc
#pragma unroll 2
  for (int kc = 0; kc < Hc / 32; ++kc) {
    v16bf a  = load_a_frag(hb_in, Hc, m0, kc * 32, lane);
    v16bf br = load_b_frag(Whr_p, kc, n_tile, n_tiles, lane);
    v16bf bz = load_b_frag(Whz_p, kc, n_tile, n_tiles, lane);
    v16bf bn = load_b_frag(Whn_p, kc, n_tile, n_tiles, lane);
    acc_r  = wmma_bf16(a, br, acc_r);
    acc_z  = wmma_bf16(a, bz, acc_z);
    acc_hn = wmma_bf16(a, bn, acc_hn);
  }

  // pointwise gates. C/D layout: lane n = n0 + L%16, VGPR j -> m = m0 + 8*(L/16) + j
  const int n    = n0 + (lane & 15);
  const int half = lane >> 4;
  const float bir = b_ir[n], biz = b_iz[n], bin = b_in[n], bhn = b_hn[n];
#pragma unroll
  for (int j = 0; j < 8; ++j) {
    const int m = m0 + half * 8 + j;
    const float hp = h_in[(size_t)m * Hc + n];
    const float r  = sigmoidf_(acc_r[j] + bir);
    const float z  = sigmoidf_(acc_z[j] + biz);
    const float nn = tanhf(acc_xn[j] + bin + r * (acc_hn[j] + bhn));
    const float hn_ = (1.0f - z) * nn + z * hp;
    h_out[(size_t)m * Hc + n]  = hn_;
    hb_out[(size_t)m * Hc + n] = f32_to_bf16(hn_);
    // streaming output: non-temporal so weights stay resident in L2
    __builtin_nontemporal_store(hn_, &y[((size_t)m * Tc + t) * Hc + n]);
  }
}

// ---------------------------------------------------------------- launch
extern "C" void kernel_launch(void* const* d_in, const int* in_sizes, int n_in,
                              void* d_out, int out_size, void* d_ws, size_t ws_size,
                              hipStream_t stream) {
  (void)in_sizes; (void)n_in; (void)out_size; (void)ws_size;

  const float* x    = (const float*)d_in[0];
  const float* W_ir = (const float*)d_in[1];
  const float* W_iz = (const float*)d_in[2];
  const float* W_in = (const float*)d_in[3];
  const float* b_ir = (const float*)d_in[4];
  const float* b_iz = (const float*)d_in[5];
  const float* b_in = (const float*)d_in[6];
  const float* W_hr = (const float*)d_in[7];
  const float* W_hz = (const float*)d_in[8];
  const float* W_hn = (const float*)d_in[9];
  const float* b_hn = (const float*)d_in[10];
  float* y = (float*)d_out;

  // workspace carve-up (~35 MB total)
  char* ws = (char*)d_ws;
  __bf16* x_bf  = (__bf16*)ws;             ws += (size_t)Bc * Tc * Ic * 2;
  __bf16* Wir_p = (__bf16*)ws;             ws += (size_t)Ic * Hc * 2;
  __bf16* Wiz_p = (__bf16*)ws;             ws += (size_t)Ic * Hc * 2;
  __bf16* Win_p = (__bf16*)ws;             ws += (size_t)Ic * Hc * 2;
  __bf16* Whr_p = (__bf16*)ws;             ws += (size_t)Hc * Hc * 2;
  __bf16* Whz_p = (__bf16*)ws;             ws += (size_t)Hc * Hc * 2;
  __bf16* Whn_p = (__bf16*)ws;             ws += (size_t)Hc * Hc * 2;
  float*  h_f32 = (float*)ws;              ws += (size_t)2 * Bc * Hc * 4;  // ping-pong
  __bf16* h_bf  = (__bf16*)ws;             ws += (size_t)2 * Bc * Hc * 2;  // ping-pong

  // convert x to bf16
  {
    size_t n = (size_t)Bc * Tc * Ic;
    cvt_f32_bf16<<<(unsigned)((n + 255) / 256), 256, 0, stream>>>(x, x_bf, n);
  }
  // pack all six weight matrices into B-fragment order
  {
    unsigned bi = (unsigned)(((size_t)Ic * Hc + 255) / 256);
    unsigned bh = (unsigned)(((size_t)Hc * Hc + 255) / 256);
    pack_w<<<bi, 256, 0, stream>>>(W_ir, Wir_p, Ic, Hc);
    pack_w<<<bi, 256, 0, stream>>>(W_iz, Wiz_p, Ic, Hc);
    pack_w<<<bi, 256, 0, stream>>>(W_in, Win_p, Ic, Hc);
    pack_w<<<bh, 256, 0, stream>>>(W_hr, Whr_p, Hc, Hc);
    pack_w<<<bh, 256, 0, stream>>>(W_hz, Whz_p, Hc, Hc);
    pack_w<<<bh, 256, 0, stream>>>(W_hn, Whn_p, Hc, Hc);
  }
  // h0 = 0 into buffer 0
  init_h<<<(Bc * Hc + 255) / 256, 256, 0, stream>>>(h_f32, h_bf, Bc * Hc);

  // sequential scan: one kernel per step, ping-pong h; kernel boundary = sync
  for (int t = 0; t < Tc; ++t) {
    const int cur = t & 1, nxt = cur ^ 1;
    gru_step<<<32, 128, 0, stream>>>(
        x_bf, Wir_p, Wiz_p, Win_p, Whr_p, Whz_p, Whn_p,
        b_ir, b_iz, b_in, b_hn,
        h_f32 + (size_t)cur * Bc * Hc, h_bf + (size_t)cur * Bc * Hc,
        h_f32 + (size_t)nxt * Bc * Hc, h_bf + (size_t)nxt * Bc * Hc,
        y, t);
  }
}
